// SelfAttentionBlock_71889162600479
// MI455X (gfx1250) — compile-verified, matrix operand order checked by reading, not run on hardware
//
#include <hip/hip_runtime.h>

#define HWK 4096
#define CK  256
#define LDT 264   // padded pitch (halves) for 256-wide LDS tiles
#define LDW 40    // padded pitch (halves) for 32-wide LDS tiles

// Flash-attention LDS layout (byte offsets into dynamic LDS):
#define KOFF0 0u
#define KOFF1 16896u      // 32 x LDT x 2
#define VOFF0 33792u
#define VOFF1 54272u      // 256 x LDW x 2
#define POFF  74752u      // 8 x 16 x LDW x 2 = 10240
#define QOFF  84992u      // 128 x LDT x 2 = 67584
#define ATTLDS 152576u

typedef __attribute__((ext_vector_type(16))) _Float16 v16h;
typedef __attribute__((ext_vector_type(8)))  _Float16 v8h;
typedef __attribute__((ext_vector_type(8)))  float    v8f;
typedef __attribute__((ext_vector_type(4)))  float    v4f;

// Load a 16x32 f16 fragment (A-layout; also used for B when source is stored
// column-major, i.e. rows of the buffer are columns of B).
// Lane l: row = l&15, k-group0 = 8*(l>>4)+0..7, k-group1 = 16+8*(l>>4)+0..7.
__device__ __forceinline__ v16h load_frag(const _Float16* base, int ld) {
  const int l = threadIdx.x & 31;
  const _Float16* p = base + (l & 15) * ld + ((l >> 4) << 3);
  v8h lo = *(const v8h*)(p);
  v8h hi = *(const v8h*)(p + 16);
  return __builtin_shufflevector(lo, hi, 0, 1, 2, 3, 4, 5, 6, 7,
                                 8, 9, 10, 11, 12, 13, 14, 15);
}

__device__ __forceinline__ v8f wmma_f16(v16h a, v16h b, v8f c) {
  return __builtin_amdgcn_wmma_f32_16x16x32_f16(
      false, a, false, b, (short)0, c, false, false);
}

// Async DMA: 16 bytes global -> LDS, no VGPR round-trip, tracked by ASYNCcnt.
__device__ __forceinline__ void async_copy_b128(unsigned lds_off,
                                                const _Float16* g) {
  asm volatile("global_load_async_to_lds_b128 %0, %1, off"
               :
               : "v"(lds_off), "v"((unsigned long long)(uintptr_t)g)
               : "memory");
}
__device__ __forceinline__ void wait_async0() {
  asm volatile("s_wait_asynccnt 0x0" ::: "memory");
}

// ---------------------------------------------------------------------------
// Kernel 1: QKV projection.  QKV_b(768x4096) = Wqkv(768x256) @ X_b(256x4096).
// Writes Qh,Kh as [b][n][c] f16, Vh as [b][c][n] f16.
// ---------------------------------------------------------------------------
__global__ __launch_bounds__(256) void qkv_proj_kernel(
    const float* __restrict__ x, const float* __restrict__ w_qkv,
    const float* __restrict__ b_qkv, _Float16* __restrict__ Qh,
    _Float16* __restrict__ Kh, _Float16* __restrict__ Vh) {
  extern __shared__ char smem[];
  _Float16* sA = (_Float16*)smem;                  // 64  x LDT (weights, f16)
  _Float16* sB = (_Float16*)(smem + 64 * LDT * 2); // 128 x LDT (x^T,    f16)

  const int tid = threadIdx.x;
  const int wave = tid >> 5;
  const int lane = tid & 31;
  const int n0 = blockIdx.x * 128;  // 32 tiles over HW
  const int o0 = blockIdx.y * 64;   // 12 tiles over 768
  const int bz = blockIdx.z;

  // Stage weight tile 64x256 (f32 -> f16), row-major.
#pragma unroll
  for (int i = 0; i < 16; ++i) {
    int flat4 = tid + 256 * i;  // 4096 x float4
    int o = flat4 >> 6;
    int c4 = (flat4 & 63) << 2;
    v4f f = *(const v4f*)(w_qkv + (size_t)(o0 + o) * CK + c4);
    _Float16* d = sA + o * LDT + c4;
    d[0] = (_Float16)f.x; d[1] = (_Float16)f.y;
    d[2] = (_Float16)f.z; d[3] = (_Float16)f.w;
  }
  // Stage X tile transposed: sB[n][c], coalesced reads along hw.
  {
    const int grp = tid >> 5;
#pragma unroll 4
    for (int c = grp; c < CK; c += 8) {
      v4f f = *(const v4f*)(x + ((size_t)bz * CK + c) * HWK + n0 + lane * 4);
      int nr = lane * 4;
      sB[(nr + 0) * LDT + c] = (_Float16)f.x;
      sB[(nr + 1) * LDT + c] = (_Float16)f.y;
      sB[(nr + 2) * LDT + c] = (_Float16)f.z;
      sB[(nr + 3) * LDT + c] = (_Float16)f.w;
    }
  }
  __syncthreads();

  const int m16 = lane & 15, h16 = lane >> 4;
  const int mt = wave & 3;          // o sub-tile
  const int ntb = (wave >> 2) * 4;  // n sub-tile base

#pragma unroll
  for (int i = 0; i < 4; ++i) {
    const int nt = ntb + i;
    v8f acc = {};
#pragma unroll
    for (int k = 0; k < 8; ++k) {
      v16h a = load_frag(sA + mt * 16 * LDT + k * 32, LDT);
      v16h bb = load_frag(sB + nt * 16 * LDT + k * 32, LDT);
      acc = wmma_f16(a, bb, acc);
    }
    const int n_g = n0 + nt * 16 + m16;
#pragma unroll
    for (int r = 0; r < 8; ++r) {
      int o_g = o0 + mt * 16 + r + 8 * h16;
      _Float16 hv = (_Float16)(acc[r] + b_qkv[o_g]);
      if (o0 < 256) {
        Qh[((size_t)bz * HWK + n_g) * CK + o_g] = hv;
      } else if (o0 < 512) {
        Kh[((size_t)bz * HWK + n_g) * CK + (o_g - 256)] = hv;
      } else {
        Vh[((size_t)bz * CK + (o_g - 512)) * HWK + n_g] = hv;
      }
    }
  }
}

// ---------------------------------------------------------------------------
// Kernel 2: flash attention.  Per workgroup: 128 query rows (16/wave).
// K/V stream through double-buffered LDS via global_load_async_to_lds_b128
// (ASYNCcnt DMA overlaps the WMMAs of the current chunk); online softmax;
// scores never touch global memory.  Output att as [b][n][c] f16.
// ---------------------------------------------------------------------------
__global__ __launch_bounds__(256) void flash_attn_kernel(
    const _Float16* __restrict__ Qh, const _Float16* __restrict__ Kh,
    const _Float16* __restrict__ Vh, _Float16* __restrict__ Ah) {
  extern __shared__ char smem[];
  _Float16* sP = (_Float16*)(smem + POFF);  // 8 waves x 16 x LDW
  _Float16* sQ = (_Float16*)(smem + QOFF);  // 128 x LDT
  const unsigned ldsbase = __builtin_amdgcn_groupstaticsize();

  const int tid = threadIdx.x;
  const int wave = tid >> 5;
  const int lane = tid & 31;
  const int bz = blockIdx.x >> 5;  // 8 batches
  const int n0 = (blockIdx.x & 31) * 128;

  const _Float16* Qb = Qh + (size_t)bz * HWK * CK;
  const _Float16* Kb = Kh + (size_t)bz * HWK * CK;
  const _Float16* Vb = Vh + (size_t)bz * CK * HWK;

  // Issue async DMA for a K/V chunk into buffer `buf` (0/1).
  auto stage_chunk = [&](int mc, int buf) {
    const unsigned kbase = buf ? KOFF1 : KOFF0;
    const unsigned vbase = buf ? VOFF1 : VOFF0;
    const int row = tid >> 3;         // K: 32 rows, 8 threads/row
    const int c0 = (tid & 7) * 32;
    const _Float16* srcK = Kb + (size_t)(mc + row) * CK + c0;
    unsigned dK = ldsbase + kbase + (unsigned)(row * LDT + c0) * 2;
#pragma unroll
    for (int i = 0; i < 4; ++i) async_copy_b128(dK + i * 16, srcK + i * 8);
    const _Float16* srcV = Vb + (size_t)tid * HWK + mc;  // V: [c][m]
    unsigned dV = ldsbase + vbase + (unsigned)(tid * LDW) * 2;
#pragma unroll
    for (int i = 0; i < 4; ++i) async_copy_b128(dV + i * 16, srcV + i * 8);
  };

  // Stage the 128x256 Q tile (async DMA) and the first K/V chunk.
  {
    const int row = tid >> 1;
    const int c0 = (tid & 1) * 128;
    const _Float16* src = Qb + (size_t)(n0 + row) * CK + c0;
    unsigned dQ = ldsbase + QOFF + (unsigned)(row * LDT + c0) * 2;
#pragma unroll
    for (int i = 0; i < 16; ++i) async_copy_b128(dQ + i * 16, src + i * 8);
  }
  stage_chunk(0, 0);
  wait_async0();
  __syncthreads();

  const int m16 = lane & 15, h16 = lane >> 4;
  v8f acc[16];  // O accumulator: 16 rows x 256 channels (fp32)
  v8f zero = {};
#pragma unroll
  for (int i = 0; i < 16; ++i) acc[i] = zero;
  float mI[8], lI[8];
#pragma unroll
  for (int r = 0; r < 8; ++r) { mI[r] = -3.0e38f; lI[r] = 0.0f; }

  _Float16* myP = sP + wave * (16 * LDW);
  int cur = 0;

  for (int mc = 0; mc < HWK; mc += 32) {
    // Kick DMA for the next chunk while we compute on the current one.
    if (mc + 32 < HWK) stage_chunk(mc + 32, cur ^ 1);
    const _Float16* sK = (const _Float16*)(smem + (cur ? KOFF1 : KOFF0));
    const _Float16* sV = (const _Float16*)(smem + (cur ? VOFF1 : VOFF0));

    // S(16x32) = Q(16x256) . K^T : two 16x16 WMMA tiles, 8 k-steps each.
    v8f s0 = zero, s1 = zero;
#pragma unroll
    for (int k = 0; k < 8; ++k) {
      v16h a = load_frag(sQ + (wave * 16) * LDT + k * 32, LDT);
      v16h b0 = load_frag(sK + 0 * LDT + k * 32, LDT);
      v16h b1 = load_frag(sK + 16 * LDT + k * 32, LDT);
      s0 = wmma_f16(a, b0, s0);
      s1 = wmma_f16(a, b1, s1);
    }

    // Online softmax update (scale = C^-0.5 = 1/16).
#pragma unroll
    for (int r = 0; r < 8; ++r) {
      float v0 = s0[r] * 0.0625f;
      float v1 = s1[r] * 0.0625f;
      float mx = fmaxf(v0, v1);
#pragma unroll
      for (int off = 8; off >= 1; off >>= 1)
        mx = fmaxf(mx, __shfl_xor(mx, off, 32));
      float nm = fmaxf(mI[r], mx);
      float alpha = __expf(mI[r] - nm);
      float p0 = __expf(v0 - nm);
      float p1 = __expf(v1 - nm);
      float rs = p0 + p1;
#pragma unroll
      for (int off = 8; off >= 1; off >>= 1) rs += __shfl_xor(rs, off, 32);
      lI[r] = lI[r] * alpha + rs;
      mI[r] = nm;
#pragma unroll
      for (int ct = 0; ct < 16; ++ct) acc[ct][r] *= alpha;
      // D-layout -> row-major P tile in per-wave LDS (A-layout source).
      myP[(r + 8 * h16) * LDW + m16] = (_Float16)p0;
      myP[(r + 8 * h16) * LDW + 16 + m16] = (_Float16)p1;
    }
    asm volatile("s_wait_dscnt 0" ::: "memory");  // P visible to own wave

    // O += P(16x32) . V(32x256): 16 WMMAs, one per channel tile.
    v16h pf = load_frag(myP, LDW);
#pragma unroll
    for (int ct = 0; ct < 16; ++ct) {
      v16h vf = load_frag(sV + (ct * 16) * LDW, LDW);
      acc[ct] = wmma_f16(pf, vf, acc[ct]);
    }

    wait_async0();    // our DMA for the next chunk has landed
    __syncthreads();  // everyone done reading cur + writing nxt
    cur ^= 1;
  }

  // Normalize and store att[b][n][c] as f16.
  _Float16* Ab = Ah + (size_t)bz * HWK * CK;
#pragma unroll
  for (int r = 0; r < 8; ++r) {
    float inv = 1.0f / lI[r];
    int n = n0 + wave * 16 + r + 8 * h16;
#pragma unroll
    for (int ct = 0; ct < 16; ++ct)
      Ab[(size_t)n * CK + ct * 16 + m16] = (_Float16)(acc[ct][r] * inv);
  }
}

// ---------------------------------------------------------------------------
// Kernel 3: output projection + bias + residual (fp32 out).
// ---------------------------------------------------------------------------
__global__ __launch_bounds__(256) void out_proj_kernel(
    const _Float16* __restrict__ Ah, const float* __restrict__ w_out,
    const float* __restrict__ b_out, const float* __restrict__ x,
    float* __restrict__ out) {
  extern __shared__ char smem[];
  _Float16* sA = (_Float16*)smem;                  // 64  x LDT (w_out)
  _Float16* sB = (_Float16*)(smem + 64 * LDT * 2); // 128 x LDT (att rows)

  const int tid = threadIdx.x;
  const int wave = tid >> 5;
  const int lane = tid & 31;
  const int n0 = blockIdx.x * 128;
  const int o0 = blockIdx.y * 64;  // 4 tiles over 256
  const int bz = blockIdx.z;

#pragma unroll
  for (int i = 0; i < 16; ++i) {
    int flat4 = tid + 256 * i;
    int o = flat4 >> 6;
    int c4 = (flat4 & 63) << 2;
    v4f f = *(const v4f*)(w_out + (size_t)(o0 + o) * CK + c4);
    _Float16* d = sA + o * LDT + c4;
    d[0] = (_Float16)f.x; d[1] = (_Float16)f.y;
    d[2] = (_Float16)f.z; d[3] = (_Float16)f.w;
  }
  {
    int row = tid >> 1;
    int c0 = (tid & 1) * 128;
    const _Float16* src = Ah + ((size_t)bz * HWK + n0 + row) * CK + c0;
    _Float16* dst = sB + row * LDT + c0;
#pragma unroll
    for (int i = 0; i < 16; ++i) ((v8h*)dst)[i] = ((const v8h*)src)[i];
  }
  __syncthreads();

  const int m16 = lane & 15, h16 = lane >> 4;
  const int mt = wave & 3;
  const int ntb = (wave >> 2) * 4;

#pragma unroll
  for (int i = 0; i < 4; ++i) {
    const int nt = ntb + i;
    v8f acc = {};
#pragma unroll
    for (int k = 0; k < 8; ++k) {
      v16h a = load_frag(sA + mt * 16 * LDT + k * 32, LDT);
      v16h bb = load_frag(sB + nt * 16 * LDT + k * 32, LDT);
      acc = wmma_f16(a, bb, acc);
    }
    const int n_g = n0 + nt * 16 + m16;
#pragma unroll
    for (int r = 0; r < 8; ++r) {
      int o_g = o0 + mt * 16 + r + 8 * h16;
      size_t idx = ((size_t)bz * CK + o_g) * HWK + n_g;
      out[idx] = acc[r] + b_out[o_g] + x[idx];
    }
  }
}

// ---------------------------------------------------------------------------
extern "C" void kernel_launch(void* const* d_in, const int* in_sizes, int n_in,
                              void* d_out, int out_size, void* d_ws,
                              size_t ws_size, hipStream_t stream) {
  const float* x = (const float*)d_in[0];
  const float* w_qkv = (const float*)d_in[1];
  const float* b_qkv = (const float*)d_in[2];
  const float* w_out = (const float*)d_in[3];
  const float* b_out = (const float*)d_in[4];
  float* out = (float*)d_out;

  const size_t per = (size_t)8 * HWK * CK;  // 8M halves = 16 MB each
  _Float16* Qh = (_Float16*)d_ws;
  _Float16* Kh = Qh + per;
  _Float16* Vh = Kh + per;
  _Float16* Ah = Vh + per;

  const int LDS_AB = (64 * LDT + 128 * LDT) * 2;  // 101376 B
  const int LDS_ATT = ATTLDS;                     // 152576 B

  (void)hipFuncSetAttribute((const void*)qkv_proj_kernel,
                            hipFuncAttributeMaxDynamicSharedMemorySize,
                            LDS_AB);
  (void)hipFuncSetAttribute((const void*)flash_attn_kernel,
                            hipFuncAttributeMaxDynamicSharedMemorySize,
                            LDS_ATT);
  (void)hipFuncSetAttribute((const void*)out_proj_kernel,
                            hipFuncAttributeMaxDynamicSharedMemorySize,
                            LDS_AB);

  qkv_proj_kernel<<<dim3(32, 12, 8), 256, LDS_AB, stream>>>(x, w_qkv, b_qkv,
                                                            Qh, Kh, Vh);
  flash_attn_kernel<<<dim3(256), 256, LDS_ATT, stream>>>(Qh, Kh, Vh, Ah);
  out_proj_kernel<<<dim3(32, 4, 8), 256, LDS_AB, stream>>>(Ah, w_out, b_out,
                                                           x, out);
}
